// GCNMBlock_309237645711
// MI455X (gfx1250) — matrix operand order repaired and checked by reading, not compiled, not run on hardware
//
#include <hip/hip_runtime.h>
#include <hip/hip_bf16.h>

typedef __attribute__((ext_vector_type(16))) __bf16 v16bf;
typedef __attribute__((ext_vector_type(8)))  float  v8f;

#define IND 128
#define HID 64
#define BN_EPS 1e-5f

// pack two floats -> one dword of two bf16 (RNE)
__device__ __forceinline__ unsigned pkbf(float a, float b) {
    unsigned ua = __builtin_bit_cast(unsigned, a);
    unsigned ub = __builtin_bit_cast(unsigned, b);
    ua = (ua + 0x7FFFu + ((ua >> 16) & 1u)) >> 16;
    ub = (ub + 0x7FFFu + ((ub >> 16) & 1u)) & 0xFFFF0000u;
    return ua | ub;
}

// ---------------- streaming f32 -> bf16 conversion (vectorized) ----------------
__global__ __launch_bounds__(256) void cvt_bf16_kernel(const float4* __restrict__ in,
                                                       uint2* __restrict__ out,
                                                       long long n4) {
    long long i = (long long)blockIdx.x * 256 + threadIdx.x;
    if (i >= n4) return;
    float4 v = in[i];
    uint2 r;
    r.x = pkbf(v.x, v.y);
    r.y = pkbf(v.z, v.w);
    out[i] = r;
}

// ---------------- degree over edge targets ----------------
__global__ __launch_bounds__(256) void deg_kernel(const long long* __restrict__ ei,
                                                  unsigned* __restrict__ deg, int E) {
    int e = blockIdx.x * 256 + threadIdx.x;
    if (e >= E) return;
    int c = (int)ei[(size_t)E + e];   // edge_index[1][e] : target
    atomicAdd(&deg[c], 1u);
}

// dinv[i] = rsqrt(deg_edges + 1)   (self loop makes deg always > 0)
__global__ __launch_bounds__(256) void dinv_kernel(const unsigned* __restrict__ deg,
                                                   float* __restrict__ dinv, int N) {
    int i = blockIdx.x * 256 + threadIdx.x;
    if (i >= N) return;
    dinv[i] = rsqrtf((float)deg[i] + 1.0f);
}

// ---------------- h = x @ W.T via WMMA bf16, fragments loaded packed ----------------
// One wave per 16-row tile; 4 col-tiles (64 cols); K=128 as 4 steps of 32.
// Fragment runs are contiguous bf16 in memory -> pure b128 loads, no conversion VALU.
struct BF16x16Bits { uint4 lo; uint4 hi; };   // 32 bytes == v16bf

__global__ __launch_bounds__(256) void gemm_wmma(const __bf16* __restrict__ xb,
                                                 const __bf16* __restrict__ Wb,
                                                 float* __restrict__ h, int N) {
    const int lane = threadIdx.x & 31;
    const int wave = threadIdx.x >> 5;
    const int rowbase = (blockIdx.x * 8 + wave) * 16;
    if (rowbase >= N) return;              // wave-uniform: EXEC stays all-ones for WMMA

    const bool hi = (lane >= 16);
    const int  m  = lane & 15;
    const int  arow = rowbase + m;
    const __bf16* xrow = xb + (size_t)((arow < N) ? arow : (N - 1)) * IND;

    v8f acc[4];
    #pragma unroll
    for (int t = 0; t < 4; ++t) acc[t] = (v8f){};

    #pragma unroll
    for (int s = 0; s < 4; ++s) {
        // A fragment: lane holds K = kA..kA+7 (elems 0..7) and kA+16..kA+23 (elems 8..15)
        const int kA = 32 * s + (hi ? 8 : 0);
        BF16x16Bits ta;
        ta.lo = *(const uint4*)(xrow + kA);
        ta.hi = *(const uint4*)(xrow + kA + 16);
        v16bf a = __builtin_bit_cast(v16bf, ta);

        #pragma unroll
        for (int t = 0; t < 4; ++t) {
            // B fragment: B[k][n] = W[n][k]; lane holds K = kB..kB+7 (elems 0..7) and kB+16..kB+23 (elems 8..15)
            const int col = 16 * t + m;
            const int kB  = 32 * s + (hi ? 8 : 0);
            const __bf16* wr = Wb + (size_t)col * IND + kB;
            BF16x16Bits tb;
            tb.lo = *(const uint4*)(wr);
            tb.hi = *(const uint4*)(wr + 16);
            v16bf b = __builtin_bit_cast(v16bf, tb);

            acc[t] = __builtin_amdgcn_wmma_f32_16x16x32_bf16(
                false, a, false, b, (short)0, acc[t], false, false);
        }
    }

    // D store: 32-bit C/D layout — VGPR v holds M = v + (hi?8:0), N-col = lane&15
    #pragma unroll
    for (int t = 0; t < 4; ++t) {
        const int col = 16 * t + m;
        #pragma unroll
        for (int v = 0; v < 8; ++v) {
            const int r = rowbase + v + (hi ? 8 : 0);
            if (r < N) h[(size_t)r * HID + col] = acc[t][v];
        }
    }
}

// ---------------- edge scatter: agg[col] += dinv[row]*dinv[col]*h[row] ----------------
__global__ __launch_bounds__(256) void scatter_kernel(const long long* __restrict__ ei,
                                                      const float* __restrict__ h,
                                                      const float* __restrict__ dinv,
                                                      float* __restrict__ agg, int E) {
    long long idx = (long long)blockIdx.x * 256 + threadIdx.x;
    long long e = idx >> 6;
    int f = (int)(idx & 63);
    if (e >= E) return;
    int r = (int)ei[e];                 // source
    int c = (int)ei[(size_t)E + e];     // target
    float nrm = dinv[r] * dinv[c];
    atomicAdd(&agg[(size_t)c * HID + f], nrm * h[(size_t)r * HID + f]);
}

// ---------------- self-loop + bias + ReLU, store pre-BN, accumulate column sums ----------------
__global__ __launch_bounds__(256) void post_kernel(const float* __restrict__ agg,
                                                   const float* __restrict__ h,
                                                   const float* __restrict__ dinv,
                                                   const float* __restrict__ bias,
                                                   float* __restrict__ out,
                                                   float* __restrict__ sums, int N) {
    __shared__ float ssum[256];
    __shared__ float ssq[256];
    const int f  = threadIdx.x & 63;
    const int ry = threadIdx.x >> 6;    // 0..3 row groups
    float s = 0.0f, s2 = 0.0f;
    for (int i = blockIdx.x * 4 + ry; i < N; i += gridDim.x * 4) {
        float d = dinv[i];
        float v = agg[(size_t)i * HID + f] + h[(size_t)i * HID + f] * d * d + bias[f];
        v = fmaxf(v, 0.0f);
        out[(size_t)i * HID + f] = v;
        s += v; s2 += v * v;
    }
    ssum[threadIdx.x] = s; ssq[threadIdx.x] = s2;
    __syncthreads();
    if (ry == 0) {
        s  = ssum[f] + ssum[64 + f] + ssum[128 + f] + ssum[192 + f];
        s2 = ssq[f]  + ssq[64 + f]  + ssq[128 + f]  + ssq[192 + f];
        atomicAdd(&sums[f], s);
        atomicAdd(&sums[64 + f], s2);
    }
}

// mean/var -> per-feature affine (scale, shift)
__global__ void bnparam_kernel(const float* __restrict__ sums,
                               const float* __restrict__ gamma,
                               const float* __restrict__ beta,
                               float* __restrict__ sc, int N) {
    int f = threadIdx.x;
    if (f >= HID) return;
    float invN  = 1.0f / (float)N;
    float mean  = sums[f] * invN;
    float var   = sums[64 + f] * invN - mean * mean;   // biased variance (jnp .var default)
    float scale = gamma[f] * rsqrtf(var + BN_EPS);
    sc[f]       = scale;
    sc[64 + f]  = beta[f] - mean * scale;
}

__global__ __launch_bounds__(256) void apply_kernel(float* __restrict__ out,
                                                    const float* __restrict__ sc,
                                                    long long total) {
    long long idx = (long long)blockIdx.x * 256 + threadIdx.x;
    if (idx >= total) return;
    int f = (int)(idx & 63);
    out[idx] = out[idx] * sc[f] + sc[64 + f];
}

extern "C" void kernel_launch(void* const* d_in, const int* in_sizes, int n_in,
                              void* d_out, int out_size, void* d_ws, size_t ws_size,
                              hipStream_t stream) {
    const float*     x     = (const float*)d_in[0];
    const long long* ei    = (const long long*)d_in[1];
    // d_in[2] = edge_attr: unused by the reference computation
    const float*     W     = (const float*)d_in[3];
    const float*     bias  = (const float*)d_in[4];
    const float*     gamma = (const float*)d_in[5];
    const float*     beta  = (const float*)d_in[6];
    float*           out   = (float*)d_out;

    const int N = in_sizes[0] / IND;
    const int E = in_sizes[1] / 2;

    // workspace carve-out (256B aligned)
    char* ws = (char*)d_ws;
    size_t off = 0;
    auto carve = [&](size_t bytes) -> void* {
        void* p = ws + off;
        off = (off + bytes + 255) & ~(size_t)255;
        return p;
    };
    unsigned* deg  = (unsigned*)carve((size_t)N * 4);
    float*    dinv = (float*)   carve((size_t)N * 4);
    float*    h    = (float*)   carve((size_t)N * HID * 4);
    float*    agg  = (float*)   carve((size_t)N * HID * 4);
    __bf16*   xb   = (__bf16*)  carve((size_t)N * IND * 2);
    __bf16*   Wb   = (__bf16*)  carve((size_t)HID * IND * 2);
    float*    sums = (float*)   carve(128 * 4);
    float*    sc   = (float*)   carve(128 * 4);
    (void)ws_size; (void)n_in; (void)out_size;

    hipMemsetAsync(deg,  0, (size_t)N * 4,        stream);
    hipMemsetAsync(agg,  0, (size_t)N * HID * 4,  stream);
    hipMemsetAsync(sums, 0, 128 * 4,              stream);

    // one-shot bf16 conversions of x and W (fully coalesced float4 -> bf16x4)
    long long x4 = (long long)N * IND / 4;
    long long w4 = (long long)HID * IND / 4;
    cvt_bf16_kernel<<<(unsigned)((x4 + 255) / 256), 256, 0, stream>>>(
        (const float4*)x, (uint2*)xb, x4);
    cvt_bf16_kernel<<<(unsigned)((w4 + 255) / 256), 256, 0, stream>>>(
        (const float4*)W, (uint2*)Wb, w4);

    deg_kernel <<<(E + 255) / 256, 256, 0, stream>>>(ei, deg, E);
    dinv_kernel<<<(N + 255) / 256, 256, 0, stream>>>(deg, dinv, N);
    gemm_wmma  <<<(N + 127) / 128, 256, 0, stream>>>(xb, Wb, h, N);

    long long sthreads = (long long)E * HID;
    scatter_kernel<<<(unsigned)((sthreads + 255) / 256), 256, 0, stream>>>(ei, h, dinv, agg, E);

    post_kernel   <<<256, 256, 0, stream>>>(agg, h, dinv, bias, out, sums, N);
    bnparam_kernel<<<1, 64, 0, stream>>>(sums, gamma, beta, sc, N);

    long long total = (long long)N * HID;
    apply_kernel<<<(unsigned)((total + 255) / 256), 256, 0, stream>>>(out, sc, total);
}
